// MultiHeadAttention_36799279792472
// MI455X (gfx1250) — matrix operand order repaired and checked by reading, not compiled, not run on hardware
//
#include <hip/hip_runtime.h>

#define BB 4
#define SS 2048
#define DD 512
#define HH 8
#define DHD 64
#define NROW (BB * SS) /* 8192 */

static constexpr float SCALE = 0.04419417382415922f; /* 1/sqrt(512) */
static constexpr float EPS   = 1e-5f;

typedef __attribute__((ext_vector_type(16))) _Float16 v16h;
typedef __attribute__((ext_vector_type(8)))  float    v8f;

// ---------------------------------------------------------------------------
// Fragment gathers from LDS, A/B in one shared contiguous K order (not the ISA table order), C/D per ISA
// A (16x32 f16): lane m=lane&15, halfword i -> K = i + 16*(lane>=16)
//   (one contiguous 32B run per lane -> compiler emits 2x ds_load_b128)
// B (32x16 f16): lane n=lane&15, halfword i -> K = i + 16*(lane>=16)
// C/D (16x16 f32): lane n=lane&15, VGPR r -> M = r + 8*(lane>=16)
// ---------------------------------------------------------------------------
__device__ __forceinline__ v16h frag_a_lds(const _Float16* lds, int row_base,
                                           int ld, int kofs) {
  const int lane = threadIdx.x & 31;
  const int m  = lane & 15;
  const int hf = (lane >> 4) & 1;
  const _Float16* p = lds + (row_base + m) * ld + kofs;
  v16h a;
#pragma unroll
  for (int i = 0; i < 16; ++i) {
    const int k = i + (hf << 4);
    a[i] = p[k];
  }
  return a;
}

// B stored as [n][k] (each output column's K-run contiguous)
__device__ __forceinline__ v16h frag_b_rowk(const _Float16* lds, int col_base,
                                            int ld, int kofs) {
  const int lane = threadIdx.x & 31;
  const int n  = lane & 15;
  const int hf = (lane >> 4) & 1;
  const _Float16* p = lds + (col_base + n) * ld + kofs + (hf << 4);
  v16h b;
#pragma unroll
  for (int i = 0; i < 16; ++i) b[i] = p[i];
  return b;
}

// B stored as [k][n] (e.g. V tile: rows = keys, cols = d)
__device__ __forceinline__ v16h frag_b_colk(const _Float16* lds, int col_base,
                                            int ld, int kofs) {
  const int lane = threadIdx.x & 31;
  const int n  = col_base + (lane & 15);
  const int hf = (lane >> 4) & 1;
  const _Float16* p = lds + (kofs + (hf << 4)) * ld + n;
  v16h b;
#pragma unroll
  for (int i = 0; i < 16; ++i) b[i] = p[i * ld];
  return b;
}

__device__ __forceinline__ v8f wmma_f16(v16h a, v16h b, v8f c) {
  return __builtin_amdgcn_wmma_f32_16x16x32_f16(false, a, false, b, (short)0,
                                                c, false, false);
}

// 64x32 f32 tile -> f16 LDS tile (stride 40), 128 threads
__device__ __forceinline__ void stage_f32_tile(_Float16* dst,
                                               const float* src, int t) {
#pragma unroll
  for (int it = 0; it < 4; ++it) {
    const int g = t + it * 128;
    const int row = g >> 3, c4 = (g & 7) << 2;
    const float4 f = *(const float4*)(src + (size_t)row * DD + c4);
    _Float16* d = dst + row * 40 + c4;
    d[0] = (_Float16)f.x; d[1] = (_Float16)f.y;
    d[2] = (_Float16)f.z; d[3] = (_Float16)f.w;
  }
}

// 64x32 f16 tile copy -> LDS (stride 40), 128 threads
__device__ __forceinline__ void stage_f16_tile(_Float16* dst,
                                               const _Float16* src, int t) {
#pragma unroll
  for (int it = 0; it < 4; ++it) {
    const int g = t + it * 128;
    const int row = g >> 3, c4 = (g & 7) << 2;
    *(uint2*)(dst + row * 40 + c4) =
        *(const uint2*)(src + (size_t)row * DD + c4);
  }
}

// 64x64 f16 tile: global -> LDS (stride 72) via CDNA5 async copies, no VGPR
// round-trip, tracked with ASYNCcnt.  256 threads, 16B per lane per issue.
__device__ __forceinline__ void async_stage_64x64(_Float16* dst,
                                                  const _Float16* src, int t) {
#pragma unroll
  for (int it = 0; it < 2; ++it) {
    const int c = t + it * 256;
    const int row = c >> 3, col = (c & 7) << 3;
    const unsigned lds_addr =
        (unsigned)(uintptr_t)(dst + row * 72 + col);  // low 32b = LDS offset
    const unsigned long long gaddr =
        (unsigned long long)(uintptr_t)(src + (size_t)row * DHD + col);
    asm volatile("global_load_async_to_lds_b128 %0, %1, off"
                 :: "v"(lds_addr), "v"(gaddr) : "memory");
  }
}

__device__ __forceinline__ void wait_async0() {
  asm volatile("s_wait_asynccnt 0x0" ::: "memory");
}

// ---------------------------------------------------------------------------
// Kernel 1: fused QKV projection.  out = x @ W^T + bias, written f16 to
// [B,H,S,DH].  Grid (128, 8, 3), block 128 (4 waves), 64x64 tiles,
// double-buffered LDS.
// ---------------------------------------------------------------------------
__global__ __launch_bounds__(128) void qkv_kernel(
    const float* __restrict__ x,
    const float* __restrict__ wq, const float* __restrict__ bq,
    const float* __restrict__ wk, const float* __restrict__ bk,
    const float* __restrict__ wv, const float* __restrict__ bv,
    _Float16* __restrict__ qo, _Float16* __restrict__ ko,
    _Float16* __restrict__ vo) {
  __shared__ _Float16 As[2][64 * 40];
  __shared__ _Float16 Bs[2][64 * 40];

  const int M0 = blockIdx.x * 64;
  const int N0 = blockIdx.y * 64;
  const float* w;
  const float* bias;
  _Float16* out;
  if (blockIdx.z == 0)      { w = wq; bias = bq; out = qo; }
  else if (blockIdx.z == 1) { w = wk; bias = bk; out = ko; }
  else                      { w = wv; bias = bv; out = vo; }

  const int t   = threadIdx.x;
  const int wid = t >> 5;

  v8f acc[4];
#pragma unroll
  for (int nt = 0; nt < 4; ++nt)
#pragma unroll
    for (int r = 0; r < 8; ++r) acc[nt][r] = 0.f;

  stage_f32_tile(As[0], x + (size_t)M0 * DD, t);
  stage_f32_tile(Bs[0], w + (size_t)N0 * DD, t);
  __syncthreads();

  for (int i = 0; i < DD / 32; ++i) {
    const int cur = i & 1;
    if (i + 1 < DD / 32) {  // prefetch next K-tile into the other buffer
      stage_f32_tile(As[cur ^ 1], x + (size_t)M0 * DD + (i + 1) * 32, t);
      stage_f32_tile(Bs[cur ^ 1], w + (size_t)N0 * DD + (i + 1) * 32, t);
    }
    const v16h a = frag_a_lds(As[cur], wid * 16, 40, 0);
#pragma unroll
    for (int nt = 0; nt < 4; ++nt) {
      acc[nt] = wmma_f16(a, frag_b_rowk(Bs[cur], nt * 16, 40, 0), acc[nt]);
    }
    __syncthreads();
  }

  const int lane  = t & 31;
  const int lanen = lane & 15;
  const int hf    = (lane >> 4) & 1;
#pragma unroll
  for (int nt = 0; nt < 4; ++nt) {
    const int col = N0 + nt * 16 + lanen;
    const float bv2 = bias[col];
    const int h = col >> 6, d = col & 63;
#pragma unroll
    for (int r = 0; r < 8; ++r) {
      const int gm = M0 + wid * 16 + (hf << 3) + r;
      const int bb = gm >> 11, s = gm & (SS - 1);
      out[((size_t)(bb * HH + h) * SS + s) * DHD + d] =
          (_Float16)(acc[nt][r] + bv2);
    }
  }
}

// ---------------------------------------------------------------------------
// Kernel 2: flash attention.  Grid (16, 8, 4), block 256 (8 waves).
// Each wave owns 16 query rows; keys streamed in 64-row blocks through
// double-buffered LDS via async global->LDS copies (ASYNCcnt).
// ---------------------------------------------------------------------------
__global__ __launch_bounds__(256) void attn_kernel(
    const _Float16* __restrict__ q, const _Float16* __restrict__ k,
    const _Float16* __restrict__ v, _Float16* __restrict__ ctx) {
  __shared__ _Float16 Qs[128 * 72];
  __shared__ _Float16 Ks[2][64 * 72];
  __shared__ _Float16 Vs[2][64 * 72];
  __shared__ _Float16 Ps[8 * 16 * 72];  // per-wave P transpose staging

  const int t     = threadIdx.x;
  const int wid   = t >> 5;
  const int lane  = t & 31;
  const int lanen = lane & 15;
  const int hf    = (lane >> 4) & 1;

  const int qb = blockIdx.x;
  const int h  = blockIdx.y;
  const int b  = blockIdx.z;
  const size_t head_off = (size_t)(b * HH + h) * SS * DHD;
  const _Float16* qp = q + head_off + (size_t)qb * 128 * DHD;
  const _Float16* kp = k + head_off;
  const _Float16* vp = v + head_off;

  // stage Q block 128x64 (f16 copy, padded stride)
#pragma unroll
  for (int it = 0; it < 4; ++it) {
    const int c = t + it * 256;
    const int row = c >> 3, col = (c & 7) << 3;
    *(uint4*)(Qs + row * 72 + col) =
        *(const uint4*)(qp + (size_t)row * DHD + col);
  }

  // prologue: async-stage first K/V block into buffer 0
  async_stage_64x64(Ks[0], kp, t);
  async_stage_64x64(Vs[0], vp, t);

  __syncthreads();
  const v16h aq0 = frag_a_lds(Qs, wid * 16, 72, 0);
  const v16h aq1 = frag_a_lds(Qs, wid * 16, 72, 32);

  float m_run[8], l_run[8];
  v8f oacc[4];
#pragma unroll
  for (int r = 0; r < 8; ++r) { m_run[r] = -1e30f; l_run[r] = 0.f; }
#pragma unroll
  for (int nt = 0; nt < 4; ++nt)
#pragma unroll
    for (int r = 0; r < 8; ++r) oacc[nt][r] = 0.f;

  _Float16* Pw = Ps + wid * 16 * 72;

  for (int i = 0; i < SS / 64; ++i) {
    const int cur = i & 1;
    // wait for this wave's async copies of buffer `cur`; the barrier then
    // both publishes buffer `cur` and proves all waves finished reading
    // buffer `cur^1`, so it is safe to refill below.
    wait_async0();
    __syncthreads();
    if (i + 1 < SS / 64) {
      async_stage_64x64(Ks[cur ^ 1], kp + (size_t)(i + 1) * 64 * DHD, t);
      async_stage_64x64(Vs[cur ^ 1], vp + (size_t)(i + 1) * 64 * DHD, t);
    }
    const _Float16* Kc = Ks[cur];
    const _Float16* Vc = Vs[cur];

    // scores: 16x64 tile per wave = 4 N-tiles x 2 K-chunks of WMMA
    v8f sacc[4];
#pragma unroll
    for (int nt = 0; nt < 4; ++nt) {
#pragma unroll
      for (int r = 0; r < 8; ++r) sacc[nt][r] = 0.f;
      sacc[nt] = wmma_f16(aq0, frag_b_rowk(Kc, nt * 16, 72, 0),  sacc[nt]);
      sacc[nt] = wmma_f16(aq1, frag_b_rowk(Kc, nt * 16, 72, 32), sacc[nt]);
#pragma unroll
      for (int r = 0; r < 8; ++r) sacc[nt][r] *= SCALE;
    }

    // online softmax; row (r + 8*hf) lives across the 16 lanes of this
    // half-wave in VGPR r -> reduce with xor masks 1,2,4,8.
#pragma unroll
    for (int r = 0; r < 8; ++r) {
      float mloc = fmaxf(fmaxf(sacc[0][r], sacc[1][r]),
                         fmaxf(sacc[2][r], sacc[3][r]));
#pragma unroll
      for (int msk = 1; msk <= 8; msk <<= 1)
        mloc = fmaxf(mloc, __shfl_xor(mloc, msk, 32));
      const float mnew = fmaxf(m_run[r], mloc);
      const float corr = __expf(m_run[r] - mnew);
      float rs = 0.f;
#pragma unroll
      for (int nt = 0; nt < 4; ++nt) {
        const float p = __expf(sacc[nt][r] - mnew);
        rs += p;
        Pw[(r + (hf << 3)) * 72 + nt * 16 + lanen] = (_Float16)p;
      }
#pragma unroll
      for (int msk = 1; msk <= 8; msk <<= 1) rs += __shfl_xor(rs, msk, 32);
      l_run[r] = l_run[r] * corr + rs;
      m_run[r] = mnew;
#pragma unroll
      for (int nt = 0; nt < 4; ++nt) oacc[nt][r] *= corr;
    }

    // same-wave LDS is in-order; compiler barrier keeps the P stores first
    asm volatile("" ::: "memory");
    const v16h ap0 = frag_a_lds(Pw, 0, 72, 0);
    const v16h ap1 = frag_a_lds(Pw, 0, 72, 32);
#pragma unroll
    for (int nt = 0; nt < 4; ++nt) {
      oacc[nt] = wmma_f16(ap0, frag_b_colk(Vc, nt * 16, 72, 0),  oacc[nt]);
      oacc[nt] = wmma_f16(ap1, frag_b_colk(Vc, nt * 16, 72, 32), oacc[nt]);
    }
  }

  // write context in [B,S,D] layout (heads re-interleaved) as f16
#pragma unroll
  for (int nt = 0; nt < 4; ++nt) {
#pragma unroll
    for (int r = 0; r < 8; ++r) {
      const int s   = qb * 128 + wid * 16 + (hf << 3) + r;
      const int col = h * DHD + nt * 16 + lanen;
      const float o = oacc[nt][r] / l_run[r];
      ctx[((size_t)(b * SS + s)) * DD + col] = (_Float16)o;
    }
  }
}

// ---------------------------------------------------------------------------
// Kernel 3: out projection + bias + residual -> f32 y (double-buffered LDS)
// ---------------------------------------------------------------------------
__global__ __launch_bounds__(128) void oproj_kernel(
    const _Float16* __restrict__ ctx, const float* __restrict__ wo,
    const float* __restrict__ bo, const float* __restrict__ x,
    float* __restrict__ y) {
  __shared__ _Float16 As[2][64 * 40];
  __shared__ _Float16 Bs[2][64 * 40];

  const int M0 = blockIdx.x * 64;
  const int N0 = blockIdx.y * 64;
  const int t  = threadIdx.x;
  const int wid = t >> 5;

  v8f acc[4];
#pragma unroll
  for (int nt = 0; nt < 4; ++nt)
#pragma unroll
    for (int r = 0; r < 8; ++r) acc[nt][r] = 0.f;

  stage_f16_tile(As[0], ctx + (size_t)M0 * DD, t);
  stage_f32_tile(Bs[0], wo + (size_t)N0 * DD, t);
  __syncthreads();

  for (int i = 0; i < DD / 32; ++i) {
    const int cur = i & 1;
    if (i + 1 < DD / 32) {
      stage_f16_tile(As[cur ^ 1], ctx + (size_t)M0 * DD + (i + 1) * 32, t);
      stage_f32_tile(Bs[cur ^ 1], wo + (size_t)N0 * DD + (i + 1) * 32, t);
    }
    const v16h a = frag_a_lds(As[cur], wid * 16, 40, 0);
#pragma unroll
    for (int nt = 0; nt < 4; ++nt) {
      acc[nt] = wmma_f16(a, frag_b_rowk(Bs[cur], nt * 16, 40, 0), acc[nt]);
    }
    __syncthreads();
  }

  const int lane  = t & 31;
  const int lanen = lane & 15;
  const int hf    = (lane >> 4) & 1;
#pragma unroll
  for (int nt = 0; nt < 4; ++nt) {
    const int col = N0 + nt * 16 + lanen;
    const float bv2 = bo[col];
#pragma unroll
    for (int r = 0; r < 8; ++r) {
      const int gm = M0 + wid * 16 + (hf << 3) + r;
      y[(size_t)gm * DD + col] = acc[nt][r] + bv2 + x[(size_t)gm * DD + col];
    }
  }
}

// ---------------------------------------------------------------------------
// Kernel 4: LayerNorm, one block per row of 512
// ---------------------------------------------------------------------------
__device__ __forceinline__ float block_sum(float v, float* sbuf, int t) {
  __syncthreads();
#pragma unroll
  for (int m = 16; m >= 1; m >>= 1) v += __shfl_xor(v, m, 32);
  if ((t & 31) == 0) sbuf[t >> 5] = v;
  __syncthreads();
  float r = sbuf[0];
#pragma unroll
  for (int i = 1; i < 8; ++i) r += sbuf[i];
  return r;
}

__global__ __launch_bounds__(256) void ln_kernel(
    const float* __restrict__ y, const float* __restrict__ gamma,
    const float* __restrict__ beta, float* __restrict__ out) {
  __shared__ float sbuf[8];
  const int row = blockIdx.x;
  const int t   = threadIdx.x;
  const float* yr = y + (size_t)row * DD;
  const float v0 = yr[t], v1 = yr[t + 256];
  const float mean = block_sum(v0 + v1, sbuf, t) * (1.f / DD);
  const float d0 = v0 - mean, d1 = v1 - mean;
  const float var = block_sum(d0 * d0 + d1 * d1, sbuf, t) * (1.f / DD);
  const float rstd = rsqrtf(var + EPS);
  out[(size_t)row * DD + t]       = d0 * rstd * gamma[t] + beta[t];
  out[(size_t)row * DD + t + 256] = d1 * rstd * gamma[t + 256] + beta[t + 256];
}

// ---------------------------------------------------------------------------
extern "C" void kernel_launch(void* const* d_in, const int* in_sizes, int n_in,
                              void* d_out, int out_size, void* d_ws,
                              size_t ws_size, hipStream_t stream) {
  (void)in_sizes; (void)n_in; (void)out_size; (void)ws_size;
  const float* x     = (const float*)d_in[0];
  const float* wq    = (const float*)d_in[1];
  const float* bq    = (const float*)d_in[2];
  const float* wk    = (const float*)d_in[3];
  const float* bk    = (const float*)d_in[4];
  const float* wv    = (const float*)d_in[5];
  const float* bv    = (const float*)d_in[6];
  const float* wo    = (const float*)d_in[7];
  const float* bo    = (const float*)d_in[8];
  const float* gamma = (const float*)d_in[9];
  const float* beta  = (const float*)d_in[10];
  float* out = (float*)d_out;

  char* ws = (char*)d_ws;
  const size_t MB = 1024 * 1024;
  _Float16* qbuf = (_Float16*)(ws);             //  8 MB
  _Float16* kbuf = (_Float16*)(ws + 8  * MB);   //  8 MB
  _Float16* vbuf = (_Float16*)(ws + 16 * MB);   //  8 MB
  _Float16* cbuf = (_Float16*)(ws + 24 * MB);   //  8 MB
  float*    ybuf = (float*)   (ws + 32 * MB);   // 16 MB

  qkv_kernel<<<dim3(NROW / 64, DD / 64, 3), 128, 0, stream>>>(
      x, wq, bq, wk, bk, wv, bv, qbuf, kbuf, vbuf);
  attn_kernel<<<dim3(SS / 128, HH, BB), 256, 0, stream>>>(qbuf, kbuf, vbuf,
                                                          cbuf);
  oproj_kernel<<<dim3(NROW / 64, DD / 64), 128, 0, stream>>>(cbuf, wo, bo, x,
                                                             ybuf);
  ln_kernel<<<NROW, 256, 0, stream>>>(ybuf, gamma, beta, out);
}
